// MultiHeadAttention_86148454023626
// MI455X (gfx1250) — compile-verified, matrix operand order checked by reading, not run on hardware
//
#include <hip/hip_runtime.h>

#define T_SEQ   2048
#define DM      512
#define NH      8
#define DH      64
#define BATCH   4
#define SROW    2052          // padded LDS row stride (floats): 16B aligned, bank-friendly

typedef __attribute__((ext_vector_type(16))) __bf16        v16bf;
typedef __attribute__((ext_vector_type(8)))  float         v8f;
typedef __attribute__((ext_vector_type(4)))  unsigned int  uint32x4;
typedef __attribute__((ext_vector_type(4)))  float         floatx4;
typedef __attribute__((ext_vector_type(2)))  float         floatx2;
typedef unsigned short u16;

union FragB16 {
  v16bf    v;
  uint32x4 q2[2];
  u16      us[16];
};

__device__ __forceinline__ u16 f2bf(float f) {
  union { float f; unsigned u; } c; c.f = f;
  unsigned u = c.u + 0x7FFFu + ((c.u >> 16) & 1u);   // round-to-nearest-even
  return (u16)(u >> 16);
}

// 16-bit A/B fragment (16x32): lane<16 holds K {kb..kb+7, kb+16..kb+23},
// kb = (lane>>4)*8.  p0 must already point at base + row*ld + k0 + kb.
__device__ __forceinline__ void frag_ld_bf16(FragB16& f, const u16* __restrict__ p0) {
  f.q2[0] = *(const uint32x4*)(p0);
  f.q2[1] = *(const uint32x4*)(p0 + 16);
}

__device__ __forceinline__ void frag_ld_lds(FragB16& f, const u16* p0) {
  f.q2[0] = *(const uint32x4*)(p0);
  f.q2[1] = *(const uint32x4*)(p0 + 16);
}

__device__ __forceinline__ v8f wmma_bf16(const FragB16& a, const FragB16& b, v8f c) {
  return __builtin_amdgcn_wmma_f32_16x16x32_bf16(false, a.v, false, b.v, (short)0, c,
                                                 false, false);
}

// ---------------------------------------------------------------------------
// Stage 1: Y = X @ W^T + b   (X fp32 (8192,512), W fp32 (512,512) row-major)
// OUT_MODE 0: bf16 head-major  (b,h,t,64)   (q, k)
// OUT_MODE 1: bf16 transposed  (b,h,64,t)   (v -> attn@v B-frags contiguous)
// X tiles are read exactly once -> non-temporal loads keep L2 for weights.
// ---------------------------------------------------------------------------
template<int OUT_MODE>
__global__ __launch_bounds__(256) void proj_qkv_kernel(
    const float* __restrict__ X, const float* __restrict__ W,
    const float* __restrict__ bias, u16* __restrict__ Y)
{
  __shared__ u16 lds_a[128 * 48];
  __shared__ u16 lds_b[64 * 48];

  const int tid  = threadIdx.x;
  const int lane = tid & 31;
  const int wid  = tid >> 5;
  const int m0   = blockIdx.x * 128;
  const int n0   = blockIdx.y * 64;
  const int kb   = (lane >> 4) << 3;
  const int lrow = lane & 15;

  const v8f vzero = {0.f,0.f,0.f,0.f,0.f,0.f,0.f,0.f};
  v8f acc[4]; for (int i = 0; i < 4; ++i) acc[i] = vzero;

  for (int k0 = 0; k0 < DM; k0 += 32) {
    #pragma unroll
    for (int j = 0; j < 4; ++j) {                 // A tile: 128x32 fp32 -> bf16
      int idx = tid + 256 * j;
      int r = idx >> 3, c = (idx & 7) * 4;
      floatx4 x = __builtin_nontemporal_load(
          (const floatx4*)(X + (size_t)(m0 + r) * DM + k0 + c));
      u16* d = &lds_a[r * 48 + c];
      d[0] = f2bf(x.x); d[1] = f2bf(x.y); d[2] = f2bf(x.z); d[3] = f2bf(x.w);
    }
    #pragma unroll
    for (int j = 0; j < 2; ++j) {                 // B tile: 64x32 (rows of W)
      int idx = tid + 256 * j;
      int r = idx >> 3, c = (idx & 7) * 4;
      floatx4 x = *(const floatx4*)(W + (size_t)(n0 + r) * DM + k0 + c);
      u16* d = &lds_b[r * 48 + c];
      d[0] = f2bf(x.x); d[1] = f2bf(x.y); d[2] = f2bf(x.z); d[3] = f2bf(x.w);
    }
    __syncthreads();

    FragB16 a;
    frag_ld_lds(a, &lds_a[(wid * 16 + lrow) * 48 + kb]);
    #pragma unroll
    for (int nt = 0; nt < 4; ++nt) {
      FragB16 b;
      frag_ld_lds(b, &lds_b[(nt * 16 + lrow) * 48 + kb]);
      acc[nt] = wmma_bf16(a, b, acc[nt]);
    }
    __syncthreads();
  }

  #pragma unroll
  for (int nt = 0; nt < 4; ++nt) {
    int gn = n0 + nt * 16 + lrow;
    float bv = bias[gn];
    int h = gn >> 6, d = gn & 63;
    #pragma unroll
    for (int i = 0; i < 8; ++i) {
      int gm = m0 + wid * 16 + ((lane >> 4) << 3) + i;
      int bIdx = gm >> 11, t = gm & (T_SEQ - 1);
      float val = acc[nt][i] + bv;
      size_t off = (OUT_MODE == 0)
          ? ((((size_t)(bIdx * NH + h) * T_SEQ + t) << 6) + d)
          : ((((size_t)(bIdx * NH + h) * DH + d) << 11) + t);
      Y[off] = f2bf(val);
    }
  }
}

// ---------------------------------------------------------------------------
// Stage 2 (fused): scores -> softmax -> attn write (single NT pass) -> ctx
// grid (T/32, B*H), 256 threads, LDS-resident 32x2048 score tile (fp32, then
// packed bf16 in place).  attn region of d_out written exactly once with
// non-temporal stores (537 MB stream must not evict q/k/vT from L2).
// ---------------------------------------------------------------------------
__global__ __launch_bounds__(256) void attn_fused_kernel(
    const u16* __restrict__ Qh, const u16* __restrict__ Kh,
    const u16* __restrict__ vT, float* __restrict__ attn,
    u16* __restrict__ ctx)
{
  extern __shared__ float s_sc[];                 // [32][SROW] fp32 scores
  const int tid  = threadIdx.x;
  const int lane = tid & 31;
  const int wid  = tid >> 5;
  const int bh   = blockIdx.y;
  const int q0   = blockIdx.x * 32;
  const int kb   = (lane >> 4) << 3;
  const int lrow = lane & 15;

  const u16* qb = Qh + (size_t)bh * T_SEQ * DH;
  const u16* kp = Kh + (size_t)bh * T_SEQ * DH;
  const v8f vzero = {0.f,0.f,0.f,0.f,0.f,0.f,0.f,0.f};

  // ---- Phase A: scores for this wave's 256-key strip ----------------------
  FragB16 aq[2][2];
  #pragma unroll
  for (int qs = 0; qs < 2; ++qs)
    #pragma unroll
    for (int kc = 0; kc < 2; ++kc)
      frag_ld_bf16(aq[qs][kc], qb + (q0 + qs * 16 + lrow) * DH + kc * 32 + kb);

  for (int kt = 0; kt < 16; ++kt) {
    const int n0 = wid * 256 + kt * 16;
    FragB16 b0, b1;
    frag_ld_bf16(b0, kp + (n0 + lrow) * DH + 0  + kb);
    frag_ld_bf16(b1, kp + (n0 + lrow) * DH + 32 + kb);
    #pragma unroll
    for (int qs = 0; qs < 2; ++qs) {
      v8f c = vzero;
      c = wmma_bf16(aq[qs][0], b0, c);
      c = wmma_bf16(aq[qs][1], b1, c);
      const int col = n0 + lrow;
      #pragma unroll
      for (int i = 0; i < 8; ++i) {
        int r = qs * 16 + ((lane >> 4) << 3) + i;
        s_sc[r * SROW + col] = c[i] * 0.125f;     // 1/sqrt(64)
      }
    }
  }
  __syncthreads();

  // ---- Phase B: per-row softmax (4 rows per wave), NT-write attn once,
  //      repack probabilities to bf16 in place in LDS -----------------------
  for (int rr = 0; rr < 4; ++rr) {
    const int r = wid * 4 + rr;
    float* srow = s_sc + r * SROW;

    float m = -3.0e38f;
    for (int j = lane; j < T_SEQ; j += 32) m = fmaxf(m, srow[j]);
    #pragma unroll
    for (int off = 16; off; off >>= 1) m = fmaxf(m, __shfl_xor(m, off, 32));

    float s = 0.f;
    for (int j = lane; j < T_SEQ; j += 32) {
      float e = __expf(srow[j] - m);
      srow[j] = e;
      s += e;
    }
    #pragma unroll
    for (int off = 16; off; off >>= 1) s += __shfl_xor(s, off, 32);
    const float inv = 1.0f / s;

    float* grow = attn + ((size_t)bh * T_SEQ + q0 + r) * T_SEQ;
    unsigned* hrow = (unsigned*)srow;             // bf16-pair alias (in place)
    for (int t = 0; t < T_SEQ / 64; ++t) {
      // write index (t*32+lane) always trails read index 2*(t*32+lane);
      // LDS ops are in-order within a wave, so in-place repack is safe.
      int j = (t * 32 + lane) * 2;
      float p0 = srow[j] * inv, p1 = srow[j + 1] * inv;
      floatx2 pv = {p0, p1};
      __builtin_nontemporal_store(pv, (floatx2*)(grow + j));
      hrow[t * 32 + lane] = (unsigned)f2bf(p0) | ((unsigned)f2bf(p1) << 16);
    }
  }
  __syncthreads();

  // ---- Phase C: ctx = probs @ v.  8 waves -> 2 q-subtiles x 4 d-tiles ----
  const int qs = wid & 1;
  const int dt = wid >> 1;
  v8f acc = vzero;
  const u16* sbase = (const u16*)s_sc + (size_t)(qs * 16 + lrow) * (SROW * 2);
  const u16* vb    = vT + (size_t)bh * DH * T_SEQ + (dt * 16 + lrow) * T_SEQ;

  for (int k0 = 0; k0 < T_SEQ; k0 += 32) {
    __builtin_prefetch(vb + k0 + 32 + kb, 0, 0);
    FragB16 a, b;
    frag_ld_lds(a, sbase + k0 + kb);
    frag_ld_bf16(b, vb + k0 + kb);
    acc = wmma_bf16(a, b, acc);
  }

  const int bIdx = bh >> 3, h = bh & 7;
  const int d = dt * 16 + lrow;
  #pragma unroll
  for (int i = 0; i < 8; ++i) {
    int t = q0 + qs * 16 + ((lane >> 4) << 3) + i;
    ctx[(size_t)(bIdx * T_SEQ + t) * DM + h * DH + d] = f2bf(acc[i]);
  }
}

// ---------------------------------------------------------------------------
// Stage 3: out = ctx @ W_o^T + b_o  -> fp32 out region of d_out (NT stores)
// ---------------------------------------------------------------------------
__global__ __launch_bounds__(256) void out_proj_kernel(
    const u16* __restrict__ Xbf, const float* __restrict__ W,
    const float* __restrict__ bias, float* __restrict__ Y)
{
  __shared__ u16 lds_a[128 * 48];
  __shared__ u16 lds_b[64 * 48];

  const int tid  = threadIdx.x;
  const int lane = tid & 31;
  const int wid  = tid >> 5;
  const int m0   = blockIdx.x * 128;
  const int n0   = blockIdx.y * 64;
  const int kb   = (lane >> 4) << 3;
  const int lrow = lane & 15;

  const v8f vzero = {0.f,0.f,0.f,0.f,0.f,0.f,0.f,0.f};
  v8f acc[4]; for (int i = 0; i < 4; ++i) acc[i] = vzero;

  for (int k0 = 0; k0 < DM; k0 += 32) {
    #pragma unroll
    for (int j = 0; j < 2; ++j) {                 // A tile: 128x32 bf16 copy
      int idx = tid + 256 * j;
      int r = idx >> 2, c = (idx & 3) * 8;
      *(uint32x4*)&lds_a[r * 48 + c] =
          *(const uint32x4*)(Xbf + (size_t)(m0 + r) * DM + k0 + c);
    }
    #pragma unroll
    for (int j = 0; j < 2; ++j) {                 // B tile: 64x32 fp32 -> bf16
      int idx = tid + 256 * j;
      int r = idx >> 3, c = (idx & 7) * 4;
      floatx4 x = *(const floatx4*)(W + (size_t)(n0 + r) * DM + k0 + c);
      u16* d = &lds_b[r * 48 + c];
      d[0] = f2bf(x.x); d[1] = f2bf(x.y); d[2] = f2bf(x.z); d[3] = f2bf(x.w);
    }
    __syncthreads();

    FragB16 a;
    frag_ld_lds(a, &lds_a[(wid * 16 + lrow) * 48 + kb]);
    #pragma unroll
    for (int nt = 0; nt < 4; ++nt) {
      FragB16 b;
      frag_ld_lds(b, &lds_b[(nt * 16 + lrow) * 48 + kb]);
      acc[nt] = wmma_bf16(a, b, acc[nt]);
    }
    __syncthreads();
  }

  #pragma unroll
  for (int nt = 0; nt < 4; ++nt) {
    int gn = n0 + nt * 16 + lrow;
    float bv = bias[gn];
    #pragma unroll
    for (int i = 0; i < 8; ++i) {
      int gm = m0 + wid * 16 + ((lane >> 4) << 3) + i;
      __builtin_nontemporal_store(acc[nt][i] + bv, Y + (size_t)gm * DM + gn);
    }
  }
}

// ---------------------------------------------------------------------------
extern "C" void kernel_launch(void* const* d_in, const int* in_sizes, int n_in,
                              void* d_out, int out_size, void* d_ws, size_t ws_size,
                              hipStream_t stream) {
  (void)in_sizes; (void)n_in; (void)out_size; (void)ws_size;
  const float* Q  = (const float*)d_in[0];
  const float* K  = (const float*)d_in[1];
  const float* V  = (const float*)d_in[2];
  const float* Wq = (const float*)d_in[3];
  const float* bq = (const float*)d_in[4];
  const float* Wk = (const float*)d_in[5];
  const float* bk = (const float*)d_in[6];
  const float* Wv = (const float*)d_in[7];
  const float* bv = (const float*)d_in[8];
  const float* Wo = (const float*)d_in[9];
  const float* bo = (const float*)d_in[10];

  float* out  = (float*)d_out;                       // (4,2048,512) fp32
  float* attn = out + (size_t)BATCH * T_SEQ * DM;    // (4,8,2048,2048) fp32

  const size_t HN = (size_t)BATCH * NH * T_SEQ * DH; // 4M bf16 elements
  u16* ws   = (u16*)d_ws;
  u16* qbuf = ws;                                    // (b,h,t,64)  bf16
  u16* kbuf = qbuf + HN;                             // (b,h,t,64)  bf16
  u16* vTb  = kbuf + HN;                             // (b,h,64,t)  bf16
  u16* ctx  = vTb  + HN;                             // (b,t,512)   bf16

  dim3 blk(256);
  dim3 gp(8192 / 128, DM / 64);                      // (64, 8)
  proj_qkv_kernel<0><<<gp, blk, 0, stream>>>(Q, Wq, bq, qbuf);
  proj_qkv_kernel<0><<<gp, blk, 0, stream>>>(K, Wk, bk, kbuf);
  proj_qkv_kernel<1><<<gp, blk, 0, stream>>>(V, Wv, bv, vTb);

  dim3 ga(T_SEQ / 32, BATCH * NH);                   // (64, 32)
  size_t smem = (size_t)32 * SROW * sizeof(float);   // 262,656 B < 320 KB
  attn_fused_kernel<<<ga, blk, smem, stream>>>(qbuf, kbuf, vTb, attn, ctx);

  out_proj_kernel<<<gp, blk, 0, stream>>>(ctx, Wo, bo, out);
}
